// LSTM_41429254537557
// MI455X (gfx1250) — matrix operand order repaired and checked by reading, not compiled, not run on hardware
//
#include <hip/hip_runtime.h>

// ---------------------------------------------------------------------------
// LSTM on MI455X (gfx1250, wave32, WMMA).
//
// Design (compile-only reasoning, see analysis):
//  - init kernel: pack Wcat[2048][1024] = [W_ih | W_hh] as bf16 (lives in L2,
//    4MB), bias[2048] = b_ih + b_hh, zero c-state and h-state.
//  - 1024 sequential step kernels (stream order = grid barrier, graph-friendly):
//      grid = 32 WGs (16-wide hidden slice each), block = 128 threads (4 waves)
//      * stage A = [x_t | h_{t-1}] (64 x 1024 bf16, 128KB) into LDS
//        (x gathered from the fp32 embedding table on the fly)
//      * wave g computes gate g: 4 row-tiles x 32 k-steps of
//        v_wmma_f32_16x16x32_bf16 (128 WMMA per wave per step)
//      * gates exchanged via 16KB LDS, cell update with v_exp/v_tanh,
//        fp32 h -> d_out (raw-view layout t*B*H + b*H + j),
//        bf16 h -> ping-pong buffer for the next step, c updated in place.
// ---------------------------------------------------------------------------

typedef __attribute__((ext_vector_type(16))) __bf16   v16bf;
typedef __attribute__((ext_vector_type(8)))  float    v8f;
typedef __attribute__((ext_vector_type(4)))  unsigned u32x4;

union Frag32B { u32x4 u[2]; v16bf bf; };   // 32 bytes = 8 VGPRs = one WMMA A/B fragment

#define LSTM_B    64
#define LSTM_T    1024
#define LSTM_H    512
#define LSTM_K    1024          // unified K = E + H
#define LSTM_G4   2048          // 4*H gate rows

__device__ __forceinline__ unsigned short f2bf(float f) {
  unsigned u = __builtin_bit_cast(unsigned, f);
  unsigned r = u + 0x7FFFu + ((u >> 16) & 1u);   // round-to-nearest-even
  return (unsigned short)(r >> 16);
}

// ---------------------------------------------------------------------------
// Init: Wcat bf16 pack, combined bias, zero c and h0.  One thread per Wcat elem.
// ---------------------------------------------------------------------------
__global__ void lstm_init(const float* __restrict__ Wih, const float* __restrict__ Whh,
                          const float* __restrict__ bih, const float* __restrict__ bhh,
                          unsigned short* __restrict__ wcat, float* __restrict__ bias,
                          float* __restrict__ cst, unsigned short* __restrict__ h0) {
  int idx = blockIdx.x * 256 + threadIdx.x;           // [0, 2048*1024)
  if (idx < LSTM_G4 * LSTM_K) {
    int row = idx >> 10, k = idx & 1023;
    float v = (k < LSTM_H) ? Wih[row * LSTM_H + k] : Whh[row * LSTM_H + (k - LSTM_H)];
    wcat[idx] = f2bf(v);
  }
  if (idx < LSTM_G4)            bias[idx] = bih[idx] + bhh[idx];
  if (idx < LSTM_B * LSTM_H) {  cst[idx] = 0.0f; h0[idx] = 0; }
}

// ---------------------------------------------------------------------------
// One LSTM timestep.  32 WGs x 128 threads.  Dynamic LDS: 128KB A + 16KB gates.
// ---------------------------------------------------------------------------
__global__ void __launch_bounds__(128)
lstm_step(const int*   __restrict__ tok,   // [B*T] flat token ids
          const float* __restrict__ emb,   // [V, E] fp32
          const unsigned short* __restrict__ wcat,  // [2048, 1024] bf16
          const float* __restrict__ bias,  // [2048]
          float* __restrict__ cst,         // [B, H] fp32 cell state
          const unsigned short* __restrict__ hin,   // [B, H] bf16
          unsigned short* __restrict__ hout,        // [B, H] bf16
          float* __restrict__ out,         // [T, B, H] fp32 (raw-view layout)
          int t) {
  extern __shared__ char smem[];
  unsigned short* A  = (unsigned short*)smem;                 // [64][1024] bf16
  float*          gl = (float*)(smem + LSTM_B * LSTM_K * 2);  // [64][64] gate exchange

  const int tid = threadIdx.x;
  const int j0  = blockIdx.x * 16;        // hidden slice

  // --- token cache (borrow gate-LDS before it is needed) --------------------
  int* tcache = (int*)gl;
  if (tid < LSTM_B) tcache[tid] = tok[t * LSTM_B + tid];
  __syncthreads();

  // --- stage x_t (gather + fp32->bf16) into A[:, 0:512] ---------------------
  for (int s = tid; s < LSTM_B * (LSTM_H / 4); s += 128) {    // 8192 float4's
    int b = s >> 7, q = s & 127;
    const float4* er = (const float4*)(emb + (size_t)tcache[b] * LSTM_H);
    float4 v = er[q];
    unsigned p0 = (unsigned)f2bf(v.x) | ((unsigned)f2bf(v.y) << 16);
    unsigned p1 = (unsigned)f2bf(v.z) | ((unsigned)f2bf(v.w) << 16);
    unsigned* d = (unsigned*)&A[b * LSTM_K + q * 4];
    d[0] = p0; d[1] = p1;
  }
  // --- stage h_{t-1} (bf16 copy) into A[:, 512:1024] ------------------------
  for (int s = tid; s < LSTM_B * (LSTM_H / 8); s += 128) {    // 4096 16B chunks
    int b = s >> 6, q = s & 63;
    u32x4 v = ((const u32x4*)hin)[s];
    *((u32x4*)&A[b * LSTM_K + LSTM_H + q * 8]) = v;
  }
  __syncthreads();

  // --- GEMM: wave g computes gate g, cols j0..j0+15, all 64 batch rows ------
  const int wv   = tid >> 5;        // gate index 0..3 (i,f,g,o)
  const int lane = tid & 31;
  const int ln   = lane & 15;
  const int hi   = lane >> 4;       // ISA A/B fragment half-split

  // B operand: lane ln holds W row (gate*512 + j0 + ln), 16 consecutive K.
  const unsigned short* wp = wcat + (size_t)(wv * LSTM_H + j0 + ln) * LSTM_K + hi * 16;
  const int abase = ln * LSTM_K + hi * 16;

  v8f acc0 = {}, acc1 = {}, acc2 = {}, acc3 = {};
  for (int k0 = 0; k0 < LSTM_K; k0 += 32) {
    Frag32B fb;
    const u32x4* pb = (const u32x4*)(wp + k0);
    fb.u[0] = pb[0]; fb.u[1] = pb[1];
#define LSTM_TILE(ACC, R)                                                     \
    {                                                                         \
      Frag32B fa;                                                             \
      int base = (R) * 16 * LSTM_K + abase + k0;                              \
      fa.u[0] = *(const u32x4*)&A[base];                                      \
      fa.u[1] = *(const u32x4*)&A[base + 8];                                  \
      ACC = __builtin_amdgcn_wmma_f32_16x16x32_bf16(false, fa.bf, false,      \
                fb.bf, (short)0, ACC, false, false);                          \
    }
    LSTM_TILE(acc0, 0)
    LSTM_TILE(acc1, 1)
    LSTM_TILE(acc2, 2)
    LSTM_TILE(acc3, 3)
#undef LSTM_TILE
  }

  // --- bias + scatter accumulators to gate-exchange LDS ---------------------
  const float bv = bias[wv * LSTM_H + j0 + ln];
#pragma unroll
  for (int i = 0; i < 8; ++i) {
    int rb = hi * 8 + i;            // row within 16x16 tile (ISA C layout)
    gl[(0 * 16 + rb) * 64 + wv * 16 + ln] = acc0[i] + bv;
    gl[(1 * 16 + rb) * 64 + wv * 16 + ln] = acc1[i] + bv;
    gl[(2 * 16 + rb) * 64 + wv * 16 + ln] = acc2[i] + bv;
    gl[(3 * 16 + rb) * 64 + wv * 16 + ln] = acc3[i] + bv;
  }
  __syncthreads();

  // --- LSTM cell: i,f,g,o -> c,h --------------------------------------------
  for (int s = tid; s < LSTM_B * 16; s += 128) {
    int b = s >> 4, n = s & 15;
    float ig = gl[b * 64 +  0 + n];
    float fg = gl[b * 64 + 16 + n];
    float gg = gl[b * 64 + 32 + n];
    float og = gl[b * 64 + 48 + n];
    int ci = b * LSTM_H + j0 + n;
    float iv = 1.0f / (1.0f + __expf(-ig));
    float fv = 1.0f / (1.0f + __expf(-fg));
    float gv = tanhf(gg);
    float ov = 1.0f / (1.0f + __expf(-og));
    float cn = fv * cst[ci] + iv * gv;
    float hv = ov * tanhf(cn);
    cst[ci] = cn;
    out[(size_t)t * (LSTM_B * LSTM_H) + ci] = hv;   // raw-view: t*B*H + b*H + j
    hout[ci] = f2bf(hv);
  }
}

// ---------------------------------------------------------------------------
extern "C" void kernel_launch(void* const* d_in, const int* in_sizes, int n_in,
                              void* d_out, int out_size, void* d_ws, size_t ws_size,
                              hipStream_t stream) {
  (void)in_sizes; (void)n_in; (void)out_size; (void)ws_size;
  const int*   tok = (const int*)  d_in[0];   // [B, T] int
  const float* emb = (const float*)d_in[1];   // [V, E]
  const float* Wih = (const float*)d_in[2];   // [4H, E]
  const float* Whh = (const float*)d_in[3];   // [4H, H]
  const float* bih = (const float*)d_in[4];   // [4H]
  const float* bhh = (const float*)d_in[5];   // [4H]
  float* out = (float*)d_out;

  // workspace layout (bytes): Wcat bf16 | bias f32 | c f32 | h ping-pong bf16
  char* ws = (char*)d_ws;
  unsigned short* wcat = (unsigned short*)ws;                         // 4,194,304
  float*          bias = (float*)(ws + 4194304);                      //     8,192
  float*          cst  = (float*)(ws + 4202496);                      //   131,072
  unsigned short* hb   = (unsigned short*)(ws + 4333568);             // 2x 65,536

  lstm_init<<<(LSTM_G4 * LSTM_K + 255) / 256, 256, 0, stream>>>(
      Wih, Whh, bih, bhh, wcat, bias, cst, hb /* hbuf0 zeroed */);

  const size_t lds = (size_t)LSTM_B * LSTM_K * 2 + (size_t)LSTM_B * 64 * 4; // 147456
  for (int t = 0; t < LSTM_T; ++t) {
    const unsigned short* hin  = hb + (t & 1) * (LSTM_B * LSTM_H);
    unsigned short*       hout = hb + ((t + 1) & 1) * (LSTM_B * LSTM_H);
    lstm_step<<<dim3(LSTM_H / 16), dim3(128), lds, stream>>>(
        tok, emb, wcat, bias, cst, hin, hout, out, t);
  }
}
